// SingleHeadAttention_41051297415741
// MI455X (gfx1250) — matrix operand order repaired and checked by reading, not compiled, not run on hardware
//
#include <hip/hip_runtime.h>
#include <stdint.h>

#define D_MODEL 1024
#define SEQ     2048
#define BATCH   4
#define MTOT    (BATCH * SEQ)   // 8192 tokens

typedef __attribute__((ext_vector_type(16))) __bf16 v16bf;
typedef __attribute__((ext_vector_type(8)))  float  v8f;
typedef unsigned int u32x4 __attribute__((ext_vector_type(4)));
typedef int          i32x8 __attribute__((ext_vector_type(8)));
typedef int          i32x4 __attribute__((ext_vector_type(4)));

union BF16Frag { uint4 u[2]; v16bf v; };

// A-fragment (16x32 bf16, M x K): per lane a contiguous run of 16 K values
// starting at k0 = kk + 16*hi. Two adjacent 16B chunks.
__device__ inline v16bf ldA16(const unsigned short* row, int k0) {
    BF16Frag f;
    f.u[0] = *(const uint4*)(row + k0);
    f.u[1] = *(const uint4*)(row + k0 + 8);
    return f.v;
}
// B-fragment (32x16 bf16, K x N): per lane a contiguous run of 16 K values
// starting at k0 = kk + 16*hi. Two adjacent 16B chunks.
__device__ inline v16bf ldB16(const unsigned short* row, int k0) {
    BF16Frag f;
    f.u[0] = *(const uint4*)(row + k0);
    f.u[1] = *(const uint4*)(row + k0 + 8);
    return f.v;
}

__device__ inline unsigned short f2bf(float x) {   // RNE f32 -> bf16
    unsigned int u = __float_as_uint(x);
    u += 0x7FFFu + ((u >> 16) & 1u);
    return (unsigned short)(u >> 16);
}

#define WMMA_BF16(a, b, c) \
    __builtin_amdgcn_wmma_f32_16x16x32_bf16(false, (a), false, (b), (short)0, (c), false, false)
#define V8F_ZERO (v8f){0.f,0.f,0.f,0.f,0.f,0.f,0.f,0.f}

// ---------------------------------------------------------------- converts
__global__ void cvt_f32_bf16(const float* __restrict__ in,
                             unsigned short* __restrict__ out, int n) {
    int i = blockIdx.x * blockDim.x + threadIdx.x;
    int stride = gridDim.x * blockDim.x;
    for (; i < n; i += stride) out[i] = f2bf(in[i]);
}

// ------------------------------------------------- projection GEMM (bf16 out)
// C[M x 1024] = A[M x 1024] * W^T ; W is (out_dim x in_dim) row-major bf16.
// mode 0: out row-major bf16 (M x 1024)   (Q, K)
// mode 1: out transposed per batch: out[b][n][s]  (V^T, shape B x D x S)
// Simple K loop (no manual pipelining): keeps live VGPRs low so the inner
// loop stays spill-free; inter-wave overlap (~8 waves/SIMD) hides latency.
__global__ void gemm_proj_bf16(const unsigned short* __restrict__ A,
                               const unsigned short* __restrict__ W,
                               unsigned short* __restrict__ out, int mode) {
    const int lane = threadIdx.x & 31;
    const int w    = threadIdx.x >> 5;
    const int hi   = lane >> 4;
    const int ln   = lane & 15;
    const int mTile = blockIdx.y * 64 + w * 16;
    const int nBase = blockIdx.x * 64;

    const unsigned short* arow = A + (size_t)(mTile + ln) * D_MODEL;
    v8f acc[4];
#pragma unroll
    for (int j = 0; j < 4; ++j) acc[j] = V8F_ZERO;

    for (int kk = 0; kk < D_MODEL; kk += 32) {
        v16bf a = ldA16(arow, kk + hi * 16);
#pragma unroll
        for (int j = 0; j < 4; ++j) {
            const unsigned short* wrow = W + (size_t)(nBase + j * 16 + ln) * D_MODEL;
            v16bf b = ldB16(wrow, kk + hi * 16);
            acc[j] = WMMA_BF16(a, b, acc[j]);
        }
    }
#pragma unroll
    for (int j = 0; j < 4; ++j) {
#pragma unroll
        for (int r = 0; r < 8; ++r) {
            int m = mTile + r + hi * 8;
            int n = nBase + j * 16 + ln;
            unsigned short v = f2bf(acc[j][r]);
            if (mode == 0) {
                out[(size_t)m * D_MODEL + n] = v;
            } else {
                int b = m >> 11, s = m & (SEQ - 1);
                out[((size_t)b * D_MODEL + n) * SEQ + s] = v;
            }
        }
    }
}

// Pipelined 16x16 score tile: C = Qs(16x1024, LDS) x K_rows(16x1024)^T
// Depth-1 prefetch of the global K fragment (~40 live VGPRs, spill-free).
__device__ inline v8f score_tile(const unsigned short* qrow,
                                 const unsigned short* krow, int hi) {
    v8f acc = V8F_ZERO;
    v16bf b_cur = ldB16(krow, hi * 16);
    int kk = 0;
    for (; kk < D_MODEL - 32; kk += 32) {
        v16bf b_nxt = ldB16(krow, kk + 32 + hi * 16);
        v16bf a = ldA16(qrow, kk + hi * 16);            // LDS (ds_load_b128)
        acc = WMMA_BF16(a, b_cur, acc);
        b_cur = b_nxt;
    }
    acc = WMMA_BF16(ldA16(qrow, kk + hi * 16), b_cur, acc);
    return acc;
}

// ----------------------------------------------------- causal attention core
// One workgroup (4 waves) per (batch, 16-query tile). Flash-style softmax.
// Q tile is staged into LDS by the Tensor Data Mover (tensor_load_to_lds)
// while the workgroup zeroes the P buffer in parallel.
__global__ void attn_kernel(const unsigned short* __restrict__ Q,
                            const unsigned short* __restrict__ K,
                            const unsigned short* __restrict__ VT,
                            unsigned short* __restrict__ CTX) {
    extern __shared__ unsigned short smem[];
    unsigned short* P  = smem;               // 16 x 2048 bf16 (64 KB)
    unsigned short* Qs = smem + 16 * SEQ;    // 16 x 1024 bf16 (32 KB)
    __shared__ float s_m[4][16];
    __shared__ float s_l[4][16];

    const int qt    = blockIdx.x;
    const int b     = blockIdx.y;
    const int qBase = qt * 16;
    const int tid   = threadIdx.x;
    const int w     = tid >> 5;
    const int lane  = tid & 31;
    const int hi    = lane >> 4;
    const int ln    = lane & 15;
    const size_t bOff = (size_t)b * SEQ * D_MODEL;

    // TDM: DMA the 16x1024 bf16 Q tile (row stride 1024 elems) into LDS.
    // D# per CDNA5 ISA 8.3/8.4: group0 = {count=1, lds_addr, global_addr,
    // type=2}; group1 = {data_size=2B, tensor_dim0=1024, tensor_dim1=16,
    // tile_dim0=1024, tile_dim1=16, dim0_stride=1024}. 2D tensor ->
    // remaining groups zero.
    if (w == 0) {
        const unsigned short* gq = Q + bOff + (size_t)qBase * D_MODEL;
        unsigned long long ga = (unsigned long long)(uintptr_t)gq;
        unsigned ldsa = (unsigned)(uintptr_t)Qs;   // low 32 bits = LDS offset
        u32x4 g0 = { 1u, ldsa, (unsigned)ga,
                     (unsigned)((ga >> 32) & 0x1FFFFFFu) | 0x80000000u };
        i32x8 g1 = { 0x00010000,            // data_size = 1 (2 bytes)
                     (int)(1024u << 16),    // tensor_dim0 = 1024 (bits 79:48 lo)
                     (int)(16u << 16),      // tensor_dim1 = 16   (bits 111:80 lo)
                     (int)(1024u << 16),    // tile_dim0 = 1024   (bits 127:112)
                     16,                    // tile_dim1 = 16     (bits 143:128)
                     1024,                  // tensor_dim0_stride = 1024
                     0, 0 };
        i32x4 g2 = {0, 0, 0, 0};
        i32x4 g3 = {0, 0, 0, 0};
        i32x8 g4 = {0, 0, 0, 0, 0, 0, 0, 0};
        __builtin_amdgcn_tensor_load_to_lds(g0, g1, g2, g3, g4, 0);
    }

    // Meanwhile every thread zeroes P (4096 uint4).
    uint4 z; z.x = z.y = z.z = z.w = 0u;
    for (int i = tid; i < 4096; i += 128) ((uint4*)P)[i] = z;

    if (w == 0) __builtin_amdgcn_s_wait_tensorcnt(0);
    __syncthreads();

    const float scale = 0.03125f;           // 1/sqrt(1024)
    const float NEG   = -3.0e38f;
    float mrun[8], lrun[8], mfin[8], linv[8];
#pragma unroll
    for (int r = 0; r < 8; ++r) { mrun[r] = NEG; lrun[r] = 0.f; }

    const unsigned short* qrow = Qs + (size_t)ln * D_MODEL;

    // ---- phase 1: online softmax statistics over causal kv tiles ----
    for (int t = w; t <= qt; t += 4) {
        const unsigned short* krow = K + bOff + (size_t)(t * 16 + ln) * D_MODEL;
        v8f acc = score_tile(qrow, krow, hi);
#pragma unroll
        for (int r = 0; r < 8; ++r) {
            int row = r + hi * 8;
            float v = acc[r] * scale;
            if (t == qt && ln > row) v = NEG;     // causal mask (diagonal tile)
            float tm = v;
            tm = fmaxf(tm, __shfl_xor(tm, 1));
            tm = fmaxf(tm, __shfl_xor(tm, 2));
            tm = fmaxf(tm, __shfl_xor(tm, 4));
            tm = fmaxf(tm, __shfl_xor(tm, 8));
            float mnew = fmaxf(mrun[r], tm);
            float e = __expf(v - mnew);
            e += __shfl_xor(e, 1);
            e += __shfl_xor(e, 2);
            e += __shfl_xor(e, 4);
            e += __shfl_xor(e, 8);
            lrun[r] = lrun[r] * __expf(mrun[r] - mnew) + e;
            mrun[r] = mnew;
        }
    }
    if (ln == 0) {
#pragma unroll
        for (int r = 0; r < 8; ++r) {
            s_m[w][r + hi * 8] = mrun[r];
            s_l[w][r + hi * 8] = lrun[r];
        }
    }
    __syncthreads();
#pragma unroll
    for (int r = 0; r < 8; ++r) {
        int row = r + hi * 8;
        float mf = s_m[0][row];
        for (int v = 1; v < 4; ++v) mf = fmaxf(mf, s_m[v][row]);
        float lf = 0.f;
        for (int v = 0; v < 4; ++v) lf += s_l[v][row] * __expf(s_m[v][row] - mf);
        mfin[r] = mf;
        linv[r] = 1.0f / lf;
    }

    // ---- phase 2: recompute scores, write normalized P (bf16) to LDS ----
    for (int t = w; t <= qt; t += 4) {
        const unsigned short* krow = K + bOff + (size_t)(t * 16 + ln) * D_MODEL;
        v8f acc = score_tile(qrow, krow, hi);
#pragma unroll
        for (int r = 0; r < 8; ++r) {
            int row = r + hi * 8;
            float v = acc[r] * scale;
            if (t == qt && ln > row) v = NEG;
            float p = __expf(v - mfin[r]) * linv[r];
            P[(size_t)row * SEQ + t * 16 + ln] = f2bf(p);
        }
    }
    __syncthreads();

    // ---- phase 3: ctx = P * V  (B frags contiguous thanks to V^T layout) ----
    const int L = ((qBase + 16) + 31) & ~31;      // kv extent, rounded to 32 (P zero-padded)
    const unsigned short* VTb = VT + (size_t)b * D_MODEL * SEQ;
    const unsigned short* prow = P + (size_t)ln * SEQ;
    for (int dt = w; dt < 64; dt += 4) {
        v8f acc = V8F_ZERO;
        const unsigned short* vrow = VTb + (size_t)(dt * 16 + ln) * SEQ;
        v16bf b_cur = ldB16(vrow, hi * 16);
        int kk = 0;
        for (; kk < L - 32; kk += 32) {
            v16bf b_nxt = ldB16(vrow, kk + 32 + hi * 16);
            v16bf a = ldA16(prow, kk + hi * 16);      // LDS
            acc = WMMA_BF16(a, b_cur, acc);
            b_cur = b_nxt;
        }
        acc = WMMA_BF16(ldA16(prow, kk + hi * 16), b_cur, acc);
#pragma unroll
        for (int r = 0; r < 8; ++r)
            CTX[bOff + (size_t)(qBase + r + hi * 8) * D_MODEL + dt * 16 + ln] = f2bf(acc[r]);
    }
}

// --------------------------------------------- output projection (fp32 + bias)
__global__ void gemm_out_f32(const unsigned short* __restrict__ A,
                             const unsigned short* __restrict__ W,
                             const float* __restrict__ bias,
                             float* __restrict__ out) {
    const int lane = threadIdx.x & 31;
    const int w    = threadIdx.x >> 5;
    const int hi   = lane >> 4;
    const int ln   = lane & 15;
    const int mTile = blockIdx.y * 64 + w * 16;
    const int nBase = blockIdx.x * 64;

    const unsigned short* arow = A + (size_t)(mTile + ln) * D_MODEL;
    v8f acc[4];
#pragma unroll
    for (int j = 0; j < 4; ++j) acc[j] = V8F_ZERO;

    for (int kk = 0; kk < D_MODEL; kk += 32) {
        v16bf a = ldA16(arow, kk + hi * 16);
#pragma unroll
        for (int j = 0; j < 4; ++j) {
            const unsigned short* wrow = W + (size_t)(nBase + j * 16 + ln) * D_MODEL;
            acc[j] = WMMA_BF16(a, ldB16(wrow, kk + hi * 16), acc[j]);
        }
    }
#pragma unroll
    for (int j = 0; j < 4; ++j) {
        int n = nBase + j * 16 + ln;
        float bb = bias[n];
#pragma unroll
        for (int r = 0; r < 8; ++r) {
            int m = mTile + r + hi * 8;
            out[(size_t)m * D_MODEL + n] = acc[j][r] + bb;
        }
    }
}

// ----------------------------------------------------------------- launcher
extern "C" void kernel_launch(void* const* d_in, const int* in_sizes, int n_in,
                              void* d_out, int out_size, void* d_ws, size_t ws_size,
                              hipStream_t stream) {
    const float* x  = (const float*)d_in[0];
    const float* Wq = (const float*)d_in[1];
    const float* Wk = (const float*)d_in[2];
    const float* Wv = (const float*)d_in[3];
    const float* Wo = (const float*)d_in[4];
    const float* bo = (const float*)d_in[5];

    unsigned short* ws  = (unsigned short*)d_ws;
    unsigned short* xbf = ws;                                       // 8192x1024
    unsigned short* wqb = xbf + (size_t)MTOT * D_MODEL;
    unsigned short* wkb = wqb + (size_t)D_MODEL * D_MODEL;
    unsigned short* wvb = wkb + (size_t)D_MODEL * D_MODEL;
    unsigned short* wob = wvb + (size_t)D_MODEL * D_MODEL;
    unsigned short* Qb  = wob + (size_t)D_MODEL * D_MODEL;          // B,S,D
    unsigned short* Kb  = Qb  + (size_t)MTOT * D_MODEL;             // B,S,D
    unsigned short* VTb = Kb  + (size_t)MTOT * D_MODEL;             // B,D,S
    unsigned short* CTX = VTb + (size_t)MTOT * D_MODEL;             // B,S,D

    cvt_f32_bf16<<<2048, 256, 0, stream>>>(x,  xbf, MTOT * D_MODEL);
    cvt_f32_bf16<<<512,  256, 0, stream>>>(Wq, wqb, D_MODEL * D_MODEL);
    cvt_f32_bf16<<<512,  256, 0, stream>>>(Wk, wkb, D_MODEL * D_MODEL);
    cvt_f32_bf16<<<512,  256, 0, stream>>>(Wv, wvb, D_MODEL * D_MODEL);
    cvt_f32_bf16<<<512,  256, 0, stream>>>(Wo, wob, D_MODEL * D_MODEL);

    dim3 gGemm(D_MODEL / 64, MTOT / 64);   // (16, 128)
    dim3 bGemm(128);
    gemm_proj_bf16<<<gGemm, bGemm, 0, stream>>>(xbf, wqb, Qb, 0);
    gemm_proj_bf16<<<gGemm, bGemm, 0, stream>>>(xbf, wkb, Kb, 0);
    gemm_proj_bf16<<<gGemm, bGemm, 0, stream>>>(xbf, wvb, VTb, 1);

    const size_t smem = (size_t)(16 * SEQ + 16 * D_MODEL) * sizeof(unsigned short); // 96 KB
    (void)hipFuncSetAttribute(reinterpret_cast<const void*>(&attn_kernel),
                              hipFuncAttributeMaxDynamicSharedMemorySize, (int)smem);
    attn_kernel<<<dim3(SEQ / 16, BATCH), 128, smem, stream>>>(Qb, Kb, VTb, CTX);

    gemm_out_f32<<<gGemm, bGemm, 0, stream>>>(CTX, wob, bo, (float*)d_out);
}
